// Llama4MoELayer_37933151158623
// MI455X (gfx1250) — matrix operand order repaired and checked by reading, not compiled, not run on hardware
//
#include <hip/hip_runtime.h>
#include <hip/hip_bf16.h>

// Llama4-style MoE: router + top-2 dispatch + per-expert SwiGLU via bf16 WMMA.
// Shapes: B=4, S=512, D=1024, I=2048, E=8, TOP_K=2, T=B*S=2048.
//
// Pipeline:
//   1) cast x / w_gate / w_up / w_down to bf16 in workspace (one pass, BW-bound)
//   2) router (wave32 per token, gate_w in LDS) -> logits + combine weights
//   3) deterministic compaction -> per-expert token lists, exact prefix offsets
//   4) stage A: h = silu(x Wg^T) * (x Wu^T), bf16 WMMA, async-to-LDS double buffer
//   5) stage B: out += w_p * h Wd^T, bf16 WMMA, fp32 atomic scatter

#define T_TOK 2048
#define DIM   1024
#define IDIM  2048
#define NEXP  8
#define NPAIR (2 * T_TOK)            // top-2 => exactly 2T pairs
#define LISTPAD 64                   // pad so tile over-reads stay in bounds
#define HROWS (NPAIR + LISTPAD)

typedef __attribute__((ext_vector_type(16))) __bf16 v16bf;
typedef __attribute__((ext_vector_type(8)))  float  v8f;

union Frag16 {
    v16bf v;
    uint4 q[2];
};

__device__ __forceinline__ unsigned short f2bf(float f) {
    unsigned int u = __float_as_uint(f);
    u += 0x7FFFu + ((u >> 16) & 1u);   // round-to-nearest-even
    return (unsigned short)(u >> 16);
}

// gfx1250 async global->LDS copy, 16 bytes per lane. Tracked by ASYNCcnt.
__device__ __forceinline__ void async_b128(unsigned ldsOff, const unsigned short* g) {
    asm volatile("global_load_async_to_lds_b128 %0, %1, off"
                 :: "v"(ldsOff), "v"((unsigned long long)(size_t)g)
                 : "memory");
}

__device__ __forceinline__ void wait_async0() {
    asm volatile("s_wait_asynccnt 0" ::: "memory");
}

// ---------------------------------------------------------------- fp32 -> bf16 cast
__global__ void cast_bf16_kernel(const float* __restrict__ src, unsigned short* __restrict__ dst) {
    size_t idx = ((size_t)blockIdx.x * blockDim.x + threadIdx.x) * 4;
    float4 f = *reinterpret_cast<const float4*>(src + idx);
    ushort4 o;
    o.x = f2bf(f.x); o.y = f2bf(f.y); o.z = f2bf(f.z); o.w = f2bf(f.w);
    *reinterpret_cast<ushort4*>(dst + idx) = o;
}

// ---------------------------------------------------------------- router
// One wave32 per token. gate_w [E,D] staged in LDS (32 KB).
__global__ void router_kernel(const float* __restrict__ x,
                              const float* __restrict__ gate_w,
                              float* __restrict__ logits_out,   // [T,E]
                              float* __restrict__ combine) {    // [T,E]
    __shared__ float sgw[NEXP * DIM];
    const int tid = threadIdx.x;
    for (int i = tid; i < NEXP * DIM; i += 256) sgw[i] = gate_w[i];
    __syncthreads();

    const int wave = tid >> 5;
    const int lane = tid & 31;
    const int token = blockIdx.x * 8 + wave;

    float acc[NEXP];
#pragma unroll
    for (int e = 0; e < NEXP; e++) acc[e] = 0.f;

    const float* xr = x + (size_t)token * DIM;
    for (int d = lane; d < DIM; d += 32) {
        float xv = xr[d];
#pragma unroll
        for (int e = 0; e < NEXP; e++) acc[e] += xv * sgw[e * DIM + d];
    }
#pragma unroll
    for (int e = 0; e < NEXP; e++)
        for (int off = 16; off > 0; off >>= 1)
            acc[e] += __shfl_xor(acc[e], off, 32);

    if (lane == 0) {
        float mx = acc[0];
#pragma unroll
        for (int e = 1; e < NEXP; e++) mx = fmaxf(mx, acc[e]);
        float p[NEXP], s = 0.f;
#pragma unroll
        for (int e = 0; e < NEXP; e++) { p[e] = __expf(acc[e] - mx); s += p[e]; }
        float inv = 1.f / s;
#pragma unroll
        for (int e = 0; e < NEXP; e++) p[e] *= inv;

        int i1 = 0;
#pragma unroll
        for (int e = 1; e < NEXP; e++) if (p[e] > p[i1]) i1 = e;
        int i2 = (i1 == 0) ? 1 : 0;
#pragma unroll
        for (int e = 0; e < NEXP; e++) if (e != i1 && p[e] > p[i2]) i2 = e;

        float wsum = p[i1] + p[i2];
#pragma unroll
        for (int e = 0; e < NEXP; e++) {
            logits_out[(size_t)token * NEXP + e] = acc[e];
            combine[(size_t)token * NEXP + e] = 0.f;
        }
        combine[(size_t)token * NEXP + i1] = p[i1] / wsum;
        combine[(size_t)token * NEXP + i2] = p[i2] / wsum;
    }
}

// ---------------------------------------------------------------- compaction
// Deterministic serial scan: per-expert token lists + exact prefix offsets.
// Pads list tail with (token 0, weight 0) so tile over-reads are benign.
__global__ void compact_kernel(const float* __restrict__ combine,
                               int* __restrict__ tokens,
                               float* __restrict__ wts,
                               int* __restrict__ eoff) {
    if (blockIdx.x == 0 && threadIdx.x == 0) {
        int off = 0;
        for (int e = 0; e < NEXP; e++) {
            eoff[e] = off;
            for (int t = 0; t < T_TOK; t++) {
                float w = combine[(size_t)t * NEXP + e];
                if (w > 0.f) { tokens[off] = t; wts[off] = w; off++; }
            }
        }
        eoff[NEXP] = off;
        for (int p = off; p < NPAIR + LISTPAD; p++) { tokens[p] = 0; wts[p] = 0.f; }
    }
}

// ---------------------------------------------------------------- stage A
// h[p, :] = silu(x Wg^T) * (x Wu^T) for each dispatched pair p of expert e.
// Block tile 64(M) x 128(N) x 32(K); 8 waves, each 16x64 (4 sub-tiles), gate+up.
// Double-buffered LDS fed by global_load_async_to_lds_b128.
__global__ void moe_gateup_kernel(const unsigned short* __restrict__ xbf,
                                  const unsigned short* __restrict__ wgb,
                                  const unsigned short* __restrict__ wub,
                                  const int* __restrict__ tokens,
                                  const int* __restrict__ eoff,
                                  unsigned short* __restrict__ hbuf) {
    const int e     = blockIdx.z;
    const int p0    = eoff[e];
    const int cnt   = eoff[e + 1] - p0;
    const int mBase = blockIdx.y * 64;
    if (mBase >= cnt) return;
    const int iBase = blockIdx.x * 128;
    const int tid = threadIdx.x;

    __shared__ __align__(64) unsigned short As [2 * 64 * 32];
    __shared__ __align__(64) unsigned short Bgs[2 * 128 * 32];
    __shared__ __align__(64) unsigned short Bus[2 * 128 * 32];

    const unsigned asOff = (unsigned)(size_t)&As[0];
    const unsigned bgOff = (unsigned)(size_t)&Bgs[0];
    const unsigned buOff = (unsigned)(size_t)&Bus[0];

    // per-thread copy slots
    const int rA = tid >> 2, qA = (tid & 3) * 8;              // A: 64 rows x 32 K
    const int rB = tid >> 1, hB = (tid & 1) * 16;             // B: 128 rows x 32 K
    const int tokA = tokens[p0 + mBase + rA];                 // padded list => valid
    const unsigned short* xrow = xbf + (size_t)tokA * DIM;
    const unsigned short* wgp = wgb + ((size_t)e * IDIM + iBase + rB) * DIM;
    const unsigned short* wup = wub + ((size_t)e * IDIM + iBase + rB) * DIM;

    const int wave = tid >> 5, lane = tid & 31;
    const int mw = wave >> 1, nw = wave & 1;
    const int lm = lane & 15, lh = lane >> 4;

    v8f accg[4], accu[4];
#pragma unroll
    for (int n = 0; n < 4; n++) {
        accg[n] = (v8f){0.f, 0.f, 0.f, 0.f, 0.f, 0.f, 0.f, 0.f};
        accu[n] = (v8f){0.f, 0.f, 0.f, 0.f, 0.f, 0.f, 0.f, 0.f};
    }

    auto issue = [&](int k0, int buf) {
        async_b128(asOff + (unsigned)(buf * 64 * 32 + rA * 32 + qA) * 2, xrow + k0 + qA);
        async_b128(bgOff + (unsigned)(buf * 128 * 32 + rB * 32 + hB) * 2,     wgp + k0 + hB);
        async_b128(bgOff + (unsigned)(buf * 128 * 32 + rB * 32 + hB + 8) * 2, wgp + k0 + hB + 8);
        async_b128(buOff + (unsigned)(buf * 128 * 32 + rB * 32 + hB) * 2,     wup + k0 + hB);
        async_b128(buOff + (unsigned)(buf * 128 * 32 + rB * 32 + hB + 8) * 2, wup + k0 + hB + 8);
    };

    issue(0, 0);
    for (int it = 0; it < DIM / 32; ++it) {
        const int buf = it & 1;
        wait_async0();
        __syncthreads();
        if (it + 1 < DIM / 32) issue((it + 1) * 32, buf ^ 1);

        const unsigned short* Ab = As  + buf * 64 * 32;
        const unsigned short* Gb = Bgs + buf * 128 * 32;
        const unsigned short* Ub = Bus + buf * 128 * 32;

        Frag16 a;
        const int arow = mw * 16 + lm;
        a.q[0] = *reinterpret_cast<const uint4*>(Ab + arow * 32 + lh * 16);
        a.q[1] = *reinterpret_cast<const uint4*>(Ab + arow * 32 + lh * 16 + 8);
#pragma unroll
        for (int ns = 0; ns < 4; ns++) {
            int bn = nw * 64 + ns * 16 + lm;
            v16bf bg = *reinterpret_cast<const v16bf*>(Gb + bn * 32 + lh * 16);
            v16bf bu = *reinterpret_cast<const v16bf*>(Ub + bn * 32 + lh * 16);
            accg[ns] = __builtin_amdgcn_wmma_f32_16x16x32_bf16(
                false, a.v, false, bg, (short)0, accg[ns], false, false);
            accu[ns] = __builtin_amdgcn_wmma_f32_16x16x32_bf16(
                false, a.v, false, bu, (short)0, accu[ns], false, false);
        }
        __syncthreads();   // all waves done reading buf before it is refilled
    }

    // epilogue: SwiGLU, store bf16 h rows (rows >= cnt discarded)
#pragma unroll
    for (int ns = 0; ns < 4; ns++) {
#pragma unroll
        for (int r = 0; r < 8; r++) {
            int lrow = mBase + mw * 16 + lh * 8 + r;
            if (lrow < cnt) {
                float g = accg[ns][r], u = accu[ns][r];
                float hval = (g / (1.f + __expf(-g))) * u;
                int col = iBase + nw * 64 + ns * 16 + lm;
                hbuf[(size_t)(p0 + lrow) * IDIM + col] = f2bf(hval);
            }
        }
    }
}

// ---------------------------------------------------------------- stage B
// out[t, :] += w_p * h[p, :] . Wd^T   (fp32 atomic accumulate across 2 experts)
__global__ void moe_down_kernel(const unsigned short* __restrict__ hbuf,
                                const unsigned short* __restrict__ wdb,
                                const int* __restrict__ tokens,
                                const float* __restrict__ wts,
                                const int* __restrict__ eoff,
                                float* __restrict__ out) {
    const int e     = blockIdx.z;
    const int p0    = eoff[e];
    const int cnt   = eoff[e + 1] - p0;
    const int mBase = blockIdx.y * 64;
    if (mBase >= cnt) return;
    const int nBase = blockIdx.x * 128;
    const int tid = threadIdx.x;

    __shared__ __align__(64) unsigned short As[2 * 64 * 32];
    __shared__ __align__(64) unsigned short Bs[2 * 128 * 32];

    const unsigned asOff = (unsigned)(size_t)&As[0];
    const unsigned bsOff = (unsigned)(size_t)&Bs[0];

    const int rA = tid >> 2, qA = (tid & 3) * 8;
    const int rB = tid >> 1, hB = (tid & 1) * 16;
    const unsigned short* hrow = hbuf + (size_t)(p0 + mBase + rA) * IDIM;   // padded buffer
    const unsigned short* wdp  = wdb + ((size_t)e * DIM + nBase + rB) * IDIM;

    const int wave = tid >> 5, lane = tid & 31;
    const int mw = wave >> 1, nw = wave & 1;
    const int lm = lane & 15, lh = lane >> 4;

    v8f acc[4];
#pragma unroll
    for (int n = 0; n < 4; n++)
        acc[n] = (v8f){0.f, 0.f, 0.f, 0.f, 0.f, 0.f, 0.f, 0.f};

    auto issue = [&](int k0, int buf) {
        async_b128(asOff + (unsigned)(buf * 64 * 32 + rA * 32 + qA) * 2, hrow + k0 + qA);
        async_b128(bsOff + (unsigned)(buf * 128 * 32 + rB * 32 + hB) * 2,     wdp + k0 + hB);
        async_b128(bsOff + (unsigned)(buf * 128 * 32 + rB * 32 + hB + 8) * 2, wdp + k0 + hB + 8);
    };

    issue(0, 0);
    for (int it = 0; it < IDIM / 32; ++it) {
        const int buf = it & 1;
        wait_async0();
        __syncthreads();
        if (it + 1 < IDIM / 32) issue((it + 1) * 32, buf ^ 1);

        const unsigned short* Ab = As + buf * 64 * 32;
        const unsigned short* Bb = Bs + buf * 128 * 32;

        Frag16 a;
        const int arow = mw * 16 + lm;
        a.q[0] = *reinterpret_cast<const uint4*>(Ab + arow * 32 + lh * 16);
        a.q[1] = *reinterpret_cast<const uint4*>(Ab + arow * 32 + lh * 16 + 8);
#pragma unroll
        for (int ns = 0; ns < 4; ns++) {
            int bn = nw * 64 + ns * 16 + lm;
            v16bf b = *reinterpret_cast<const v16bf*>(Bb + bn * 32 + lh * 16);
            acc[ns] = __builtin_amdgcn_wmma_f32_16x16x32_bf16(
                false, a.v, false, b, (short)0, acc[ns], false, false);
        }
        __syncthreads();
    }

#pragma unroll
    for (int ns = 0; ns < 4; ns++) {
#pragma unroll
        for (int r = 0; r < 8; r++) {
            int lrow = mBase + mw * 16 + lh * 8 + r;
            if (lrow < cnt) {
                int prow = p0 + lrow;
                int tok = tokens[prow];
                float wgt = wts[prow];
                int col = nBase + nw * 64 + ns * 16 + lm;
                atomicAdd(out + (size_t)tok * DIM + col, acc[ns][r] * wgt);
            }
        }
    }
}

// ---------------------------------------------------------------- launch
extern "C" void kernel_launch(void* const* d_in, const int* in_sizes, int n_in,
                              void* d_out, int out_size, void* d_ws, size_t ws_size,
                              hipStream_t stream) {
    (void)in_sizes; (void)n_in; (void)out_size; (void)ws_size;

    const float* x      = (const float*)d_in[0];   // [B,S,D]
    const float* gate_w = (const float*)d_in[1];   // [E,D]
    const float* w_gate = (const float*)d_in[2];   // [E,I,D]
    const float* w_up   = (const float*)d_in[3];   // [E,I,D]
    const float* w_down = (const float*)d_in[4];   // [E,D,I]

    float* out    = (float*)d_out;                 // [T,D] then [T,E] logits
    float* logits = out + (size_t)T_TOK * DIM;

    const size_t WSZ = (size_t)NEXP * IDIM * DIM;  // elements per weight tensor

    char* w = (char*)d_ws;
    unsigned short* xbf = (unsigned short*)w;  w += (size_t)T_TOK * DIM * 2;
    unsigned short* wgb = (unsigned short*)w;  w += WSZ * 2;
    unsigned short* wub = (unsigned short*)w;  w += WSZ * 2;
    unsigned short* wdb = (unsigned short*)w;  w += WSZ * 2;
    float* combine      = (float*)w;           w += (size_t)T_TOK * NEXP * 4;
    int*   tokens       = (int*)w;             w += (size_t)(NPAIR + LISTPAD) * 4;
    float* wts          = (float*)w;           w += (size_t)(NPAIR + LISTPAD) * 4;
    int*   eoff         = (int*)w;             w += 128;
    unsigned short* hbuf = (unsigned short*)w; // HROWS * IDIM * 2

    hipMemsetAsync(d_out, 0, (size_t)T_TOK * DIM * sizeof(float), stream);

    cast_bf16_kernel<<<(T_TOK * DIM) / 1024, 256, 0, stream>>>(x, xbf);
    cast_bf16_kernel<<<(unsigned)(WSZ / 1024), 256, 0, stream>>>(w_gate, wgb);
    cast_bf16_kernel<<<(unsigned)(WSZ / 1024), 256, 0, stream>>>(w_up, wub);
    cast_bf16_kernel<<<(unsigned)(WSZ / 1024), 256, 0, stream>>>(w_down, wdb);

    router_kernel<<<T_TOK / 8, 256, 0, stream>>>(x, gate_w, logits, combine);
    compact_kernel<<<1, 32, 0, stream>>>(combine, tokens, wts, eoff);

    dim3 gA(IDIM / 128, T_TOK / 64, NEXP);
    moe_gateup_kernel<<<gA, 256, 0, stream>>>(xbf, wgb, wub, tokens, eoff, hbuf);

    dim3 gB(DIM / 128, T_TOK / 64, NEXP);
    moe_down_kernel<<<gB, 256, 0, stream>>>(hbuf, wdb, tokens, wts, eoff, out);
}
